// CustomSelfAttention_10660108829070
// MI455X (gfx1250) — matrix operand order repaired and checked by reading, compiled, not run on hardware
//
#include <hip/hip_runtime.h>
#include <hip/hip_bf16.h>

// Problem constants (fixed by the reference).
#define BB 4
#define SS 2048
#define HH 2048
#define NHH 16
#define HDD 128

typedef _Float16 v16h __attribute__((ext_vector_type(16)));
typedef float    v8f  __attribute__((ext_vector_type(8)));
typedef unsigned int u32x4 __attribute__((ext_vector_type(4)));

union Frag16 { v16h h; u32x4 u[2]; };

static __device__ __forceinline__ v8f wmma_f16(const Frag16& a, const Frag16& b, v8f c) {
    return __builtin_amdgcn_wmma_f32_16x16x32_f16(
        /*neg_a=*/false, a.h, /*neg_b=*/false, b.h,
        /*c_mod=*/(short)0, c, /*reuse_a=*/false, /*reuse_b=*/false);
}

__device__ __forceinline__ float hmax16(float v) {
#pragma unroll
    for (int m = 1; m < 16; m <<= 1) v = fmaxf(v, __shfl_xor(v, m, 32));
    return v;
}

// ---------------------------------------------------------------------------
// 1) f32 -> f16 elementwise convert (for x)
// ---------------------------------------------------------------------------
__global__ void cvt_f32_f16(const float* __restrict__ in, _Float16* __restrict__ out, int n) {
    int i = blockIdx.x * blockDim.x + threadIdx.x;
    if (i < n) out[i] = (_Float16)in[i];
}

// ---------------------------------------------------------------------------
// 2) Weight transpose + convert: Wt[k][n] = (f16) W[n][k]   (H x H)
// ---------------------------------------------------------------------------
__global__ void wtrans(const float* __restrict__ W, _Float16* __restrict__ Wt) {
    __shared__ _Float16 t[32][33];
    const int bx = blockIdx.x * 32;   // k tile
    const int by = blockIdx.y * 32;   // n tile
    const int tx = threadIdx.x, ty = threadIdx.y;   // block (32, 8)
#pragma unroll
    for (int i = 0; i < 32; i += 8)
        t[ty + i][tx] = (_Float16)W[(size_t)(by + ty + i) * HH + bx + tx];
    __syncthreads();
#pragma unroll
    for (int i = 0; i < 32; i += 8)
        Wt[(size_t)(bx + ty + i) * HH + by + tx] = t[tx][ty + i];
}

// ---------------------------------------------------------------------------
// 3) fp16 GEMM  C[M,N] = A[M,K] @ Bt[K,N]   (Bt is W^T stored row-major)
//    MODE 0: f16 output, natural [M][N]
//    MODE 1: f16 output, per-head transposed Kt[b][h][d][s] (packed b128 stores)
//    MODE 2: f32 output, natural [M][N]
//    Block = 256 threads (8 waves), wave tile 32x64, WG tile 128x128.
// ---------------------------------------------------------------------------
template <int MODE>
__global__ __launch_bounds__(256, 1)
void gemm16(const _Float16* __restrict__ A, const _Float16* __restrict__ Bt,
            void* __restrict__ Cv, int M, int N, int K) {
    const int lane = threadIdx.x & 31;
    const int wave = threadIdx.x >> 5;
    const int wm = wave & 3, wn = wave >> 2;           // 4 waves in M, 2 in N
    const int m0 = blockIdx.y * 128 + wm * 32;
    const int n0 = blockIdx.x * 128 + wn * 64;
    const int half = lane >> 4, l15 = lane & 15;

    v8f acc[2][4] = {};

    for (int k0 = 0; k0 < K; k0 += 32) {
        Frag16 a[2], b[4];
#pragma unroll
        for (int mi = 0; mi < 2; mi++) {
            // A fragment 16x32: lane l15 = row M, halves split K {0..7,16..23}/{8..15,24..31}
            const _Float16* p = A + (size_t)(m0 + mi * 16 + l15) * K + k0 + half * 8;
            a[mi].u[0] = *(const u32x4*)(p);
            a[mi].u[1] = *(const u32x4*)(p + 16);
        }
#pragma unroll
        for (int ni = 0; ni < 4; ni++) {
            // B fragment 32x16: lane l15 = N col, halves split K {0..7,16..23}/{8..15,24..31}
            const _Float16* p = Bt + (size_t)(k0 + half * 8) * N + n0 + ni * 16 + l15;
#pragma unroll
            for (int i = 0; i < 8; i++) {
                b[ni].h[i] = p[(size_t)i * N];
                b[ni].h[8 + i] = p[(size_t)(16 + i) * N];
            }
        }
#pragma unroll
        for (int mi = 0; mi < 2; mi++)
#pragma unroll
            for (int ni = 0; ni < 4; ni++)
                acc[mi][ni] = wmma_f16(a[mi], b[ni], acc[mi][ni]);
    }

    // C/D layout: lane&15 = N col, rows M = (lane>>4)*8 + r across the 8 VGPRs.
    if (MODE == 1) {
        // K output, per-head transposed: Kt[((b*NH+h)*HD+d)*S + s].
        // The 8 accumulator rows per lane are contiguous in s -> one b128 store.
#pragma unroll
        for (int mi = 0; mi < 2; mi++) {
#pragma unroll
            for (int ni = 0; ni < 4; ni++) {
                const int col = n0 + ni * 16 + l15;
                const int h = col >> 7, d = col & (HDD - 1);
                const int row0 = m0 + mi * 16 + half * 8;
                const int bb = row0 >> 11, s0 = row0 & (SS - 1);
                union { _Float16 e[8]; u32x4 u; } pk;
#pragma unroll
                for (int r = 0; r < 8; r++) pk.e[r] = (_Float16)acc[mi][ni][r];
                *(u32x4*)&((_Float16*)Cv)[((((size_t)bb * NHH + h) * HDD + d) * SS) + s0] = pk.u;
            }
        }
    } else {
#pragma unroll
        for (int mi = 0; mi < 2; mi++) {
#pragma unroll
            for (int ni = 0; ni < 4; ni++) {
                const int col = n0 + ni * 16 + l15;
#pragma unroll
                for (int r = 0; r < 8; r++) {
                    const int row = m0 + mi * 16 + half * 8 + r;
                    const float v = acc[mi][ni][r];
                    if (MODE == 2) ((float*)Cv)[(size_t)row * N + col] = v;
                    else           ((_Float16*)Cv)[(size_t)row * N + col] = (_Float16)v;
                }
            }
        }
    }
}

// ---------------------------------------------------------------------------
// 4) Causal flash attention.
//    Block = 128 threads (4 waves); wave owns 16 query rows; grid (S/64, NH, B).
//    Q  : f16 [B][S][H]      (head-major columns)
//    Kt : f16 [B][NH][HD][S] (pre-transposed per head)
//    V  : f16 [B][S][H]
//    Oa : f16 [B][S][H]
//    - Causal-mask block peeled: steady-state loop has no compare/select work.
//    - Row sums computed as P x ones via WMMA (broadcast in C-layout), so only
//      the row-max butterfly uses the DS/permute pipe.
// ---------------------------------------------------------------------------
__global__ __launch_bounds__(128, 1)
void flash_attn(const _Float16* __restrict__ Q, const _Float16* __restrict__ Kt,
                const _Float16* __restrict__ V, _Float16* __restrict__ Oa) {
    __shared__ _Float16 lds[4][16][32];   // per-wave private staging for P tiles

    const int lane = threadIdx.x & 31;
    const int wave = threadIdx.x >> 5;
    const int half = lane >> 4, l15 = lane & 15;
    const int b = blockIdx.z, h = blockIdx.y;
    const int q0 = (blockIdx.x * 4 + wave) * 16;

    const _Float16* Qp = Q + ((size_t)b * SS + q0) * HH + h * HDD;
    const _Float16* Kp = Kt + ((size_t)(b * NHH + h)) * HDD * SS;
    const _Float16* Vp = V + (size_t)b * SS * HH + h * HDD;

    // Q fragments: 4 chunks of K=32 over head dim (d = 0..127).
    // Fold the softmax scale (1/sqrt(128)) into Q once -> v_pk_mul_f16.
    Frag16 qf[4];
#pragma unroll
    for (int c = 0; c < 4; c++) {
        const _Float16* p = Qp + (size_t)l15 * HH + c * 32 + half * 8;
        qf[c].u[0] = *(const u32x4*)(p);
        qf[c].u[1] = *(const u32x4*)(p + 16);
        qf[c].h *= (_Float16)0.08838834764831845f;
    }

    // All-ones B fragment: one WMMA with it computes row sums of P,
    // broadcast across all 16 lanes in C-layout.
    Frag16 onesf;
#pragma unroll
    for (int i = 0; i < 16; i++) onesf.h[i] = (_Float16)1.0f;

    v8f o[8] = {};     // O accumulator: 16 rows x 128 cols
    v8f lacc = {};     // row-sum accumulator (normalizer l), same row layout
    float mrow[8];
#pragma unroll
    for (int r = 0; r < 8; r++) mrow[r] = -__builtin_inff();

    // One key-block step; MASK is a literal at every call site so the causal
    // compare/select path is compiled out of the unmasked steady-state loop.
    auto kblock = [&](int kk0, bool MASK) __attribute__((always_inline)) {
        // ---- S = (Q*scale) K^T over this 32-key block (two 16-col subtiles) ----
        v8f s[2] = {};
#pragma unroll
        for (int ns = 0; ns < 2; ns++) {
#pragma unroll
            for (int c = 0; c < 4; c++) {
                Frag16 kf;
                const _Float16* p = Kp + (size_t)(c * 32 + half * 8) * SS + kk0 + ns * 16 + l15;
#pragma unroll
                for (int i = 0; i < 8; i++) {
                    kf.h[i] = p[(size_t)i * SS];
                    kf.h[8 + i] = p[(size_t)(16 + i) * SS];
                }
                s[ns] = wmma_f16(qf[c], kf, s[ns]);
            }
        }

        // ---- (optional) causal mask + online softmax rescale ----
#pragma unroll
        for (int r = 0; r < 8; r++) {
            float t0 = s[0][r];
            float t1 = s[1][r];
            if (MASK) {
                const int qrow = q0 + half * 8 + r;
                if (kk0 + l15 > qrow)      t0 = -__builtin_inff();
                if (kk0 + 16 + l15 > qrow) t1 = -__builtin_inff();
            }
            const float mr = fmaxf(mrow[r], hmax16(fmaxf(t0, t1)));
            const float alpha = __expf(mrow[r] - mr);
            mrow[r] = mr;
            s[0][r] = __expf(t0 - mr);
            s[1][r] = __expf(t1 - mr);
            lacc[r] *= alpha;
#pragma unroll
            for (int ds = 0; ds < 8; ds++) o[ds][r] *= alpha;
        }

        // ---- P (f32 C-layout) -> f16 A-fragment via per-wave LDS staging ----
#pragma unroll
        for (int r = 0; r < 8; r++) {
            lds[wave][half * 8 + r][l15]      = (_Float16)s[0][r];
            lds[wave][half * 8 + r][16 + l15] = (_Float16)s[1][r];
        }
        asm volatile("s_wait_dscnt 0" ::: "memory");   // in-wave LDS write->read fence
        Frag16 pa;
        {
            const _Float16* p = &lds[wave][l15][half * 8];
            pa.u[0] = *(const u32x4*)(p);
            pa.u[1] = *(const u32x4*)(p + 16);
        }

        // ---- l += rowsum(P) via matrix pipe (P x ones) ----
        lacc = wmma_f16(pa, onesf, lacc);

        // ---- O += P V ----
#pragma unroll
        for (int ds = 0; ds < 8; ds++) {
            Frag16 vf;
            const _Float16* p = Vp + (size_t)(kk0 + half * 8) * HH + ds * 16 + l15;
#pragma unroll
            for (int i = 0; i < 8; i++) {
                vf.h[i] = p[(size_t)i * HH];
                vf.h[8 + i] = p[(size_t)(16 + i) * HH];
            }
            o[ds] = wmma_f16(pa, vf, o[ds]);
        }
    };

    const int nblk = (q0 >> 5) + 1;   // 32-wide key blocks touching causal region
    for (int j = 0; j < nblk - 1; j++)   // full blocks: no masking
        kblock(j * 32, false);
    kblock((nblk - 1) * 32, true);       // final (diagonal) block: masked

    // ---- epilogue: O /= l, store f16 ----
    _Float16* Op = Oa + ((size_t)b * SS + q0) * HH + h * HDD;
#pragma unroll
    for (int r = 0; r < 8; r++) {
        const float inv = 1.0f / lacc[r];
        const int row = half * 8 + r;
#pragma unroll
        for (int ds = 0; ds < 8; ds++)
            Op[(size_t)row * HH + ds * 16 + l15] = (_Float16)(o[ds][r] * inv);
    }
}

// ---------------------------------------------------------------------------
// Launch
// ---------------------------------------------------------------------------
extern "C" void kernel_launch(void* const* d_in, const int* in_sizes, int n_in,
                              void* d_out, int out_size, void* d_ws, size_t ws_size,
                              hipStream_t stream) {
    const float* x  = (const float*)d_in[0];
    const float* wq = (const float*)d_in[1];
    const float* wk = (const float*)d_in[2];
    const float* wv = (const float*)d_in[3];
    const float* wo = (const float*)d_in[4];

    const size_t ACT = (size_t)BB * SS * HH;   // 16,777,216 elements
    const size_t WEL = (size_t)HH * HH;        //  4,194,304 elements

    _Float16* xh  = (_Float16*)d_ws;
    _Float16* Qh  = xh + ACT;
    _Float16* Ktb = Qh + ACT;
    _Float16* Vh  = Ktb + ACT;
    _Float16* Ah  = Vh + ACT;
    _Float16* wqt = Ah + ACT;
    _Float16* wkt = wqt + WEL;
    _Float16* wvt = wkt + WEL;
    _Float16* wot = wvt + WEL;
    // total ws: (5*ACT + 4*WEL) * 2 bytes = 192 MiB

    // 1) convert x
    cvt_f32_f16<<<(int)(ACT / 256), 256, 0, stream>>>(x, xh, (int)ACT);

    // 2) transpose+convert weights
    dim3 tg(HH / 32, HH / 32), tb(32, 8);
    wtrans<<<tg, tb, 0, stream>>>(wq, wqt);
    wtrans<<<tg, tb, 0, stream>>>(wk, wkt);
    wtrans<<<tg, tb, 0, stream>>>(wv, wvt);
    wtrans<<<tg, tb, 0, stream>>>(wo, wot);

    // 3) QKV projections: M = B*S = 8192, N = H = 2048, K = H = 2048
    const int M = BB * SS, N = HH, K = HH;
    dim3 gg(N / 128, M / 128);
    gemm16<0><<<gg, 256, 0, stream>>>(xh, wqt, Qh, M, N, K);   // Q natural
    gemm16<1><<<gg, 256, 0, stream>>>(xh, wkt, Ktb, M, N, K);  // K per-head transposed
    gemm16<0><<<gg, 256, 0, stream>>>(xh, wvt, Vh, M, N, K);   // V natural

    // 4) causal flash attention
    flash_attn<<<dim3(SS / 64, NHH, BB), 128, 0, stream>>>(Qh, Ktb, Vh, Ah);

    // 5) output projection -> f32 d_out
    gemm16<2><<<gg, 256, 0, stream>>>(Ah, wot, (float*)d_out, M, N, K);
}
